// BaseViTSelfAttention_69922067579397
// MI455X (gfx1250) — compile-verified, matrix operand order checked by reading, not run on hardware
//
#include <hip/hip_runtime.h>
#include <hip/hip_bf16.h>

// ---------------------------------------------------------------------------
// BaseViTSelfAttention on MI455X (gfx1250): bf16 WMMA everywhere.
//   hidden  [8,1024,1024] f32,  context [8,256,1024] f32
//   Wq/Wk/Wv [1024,1024] f32, biases [1024] f32
//   out [8,1024,1024] f32
// H=16 heads, d=64. nq=1024, nk=1280.
// ---------------------------------------------------------------------------

typedef __attribute__((ext_vector_type(16))) __bf16 v16bf;
typedef __attribute__((ext_vector_type(8)))  float  v8f;

union Frag {                 // one WMMA 16x32 bf16 operand = 8 VGPRs
    v16bf v;
    uint4 u[2];
};

__device__ __forceinline__ unsigned short f2bf(float f) {
    union { float f; unsigned int u; } a; a.f = f;
    unsigned int u = a.u;
    return (unsigned short)((u + 0x7FFFu + ((u >> 16) & 1u)) >> 16);  // RNE
}

// ---------------------------------------------------------------------------
// Pack X = bf16(concat(hidden, context)) : [8][1280][1024]
// ---------------------------------------------------------------------------
__global__ __launch_bounds__(256) void pack_x_kernel(
    const float* __restrict__ hid, const float* __restrict__ ctx,
    unsigned short* __restrict__ X)
{
    size_t i = (size_t)blockIdx.x * 256 + threadIdx.x;   // 8*1280*1024 total
    size_t b   = i >> 20;              // / (1280*1024)? no: use explicit math below
    // explicit decompose: i = ((b*1280)+row)*1024 + col
    size_t bi  = i / (1280u * 1024u);
    size_t rem = i - bi * (1280u * 1024u);
    size_t row = rem >> 10;
    size_t col = rem & 1023u;
    (void)b;
    float v = (row < 1024u)
        ? hid[(bi * 1024u + row) * 1024u + col]
        : ctx[(bi * 256u + (row - 1024u)) * 1024u + col];
    X[i] = f2bf(v);
}

// ---------------------------------------------------------------------------
// Pack Wb = bf16([Wq;Wk;Wv]) : [3][1024][1024] (row n holds K contiguous)
// ---------------------------------------------------------------------------
__global__ __launch_bounds__(256) void pack_w_kernel(
    const float* __restrict__ Wq, const float* __restrict__ Wk,
    const float* __restrict__ Wv, unsigned short* __restrict__ Wb)
{
    size_t i = (size_t)blockIdx.x * 256 + threadIdx.x;   // 3*1024*1024 total
    size_t m   = i / (1024u * 1024u);
    size_t rem = i - m * (1024u * 1024u);
    const float* W = (m == 0) ? Wq : (m == 1) ? Wk : Wv;
    Wb[i] = f2bf(W[rem]);
}

// ---------------------------------------------------------------------------
// Projection GEMM: one wave computes a 16x64 tile of (X @ W^T + bias).
//   grid = (640 row-tiles, 16 col-groups, 3 matrices), block = 32 (1 wave)
//   mat 0 -> Q  [10240,1024] bf16 row-major (rows are b*1280+seq)
//   mat 1 -> K  [10240,1024] bf16 row-major
//   mat 2 -> Vt [8][16][64][1280] bf16 (transposed: dv-major, key contiguous)
// ---------------------------------------------------------------------------
__global__ __launch_bounds__(32) void proj_kernel(
    const unsigned short* __restrict__ X,
    const unsigned short* __restrict__ Wb,
    const float* __restrict__ bq, const float* __restrict__ bk,
    const float* __restrict__ bv,
    unsigned short* __restrict__ Qb, unsigned short* __restrict__ Kb,
    unsigned short* __restrict__ Vt)
{
    const int lane = threadIdx.x;
    const int r    = lane & 15;          // row (A) / column (B,D) within tile
    const int hi   = lane >> 4;          // half-wave select
    const int koff = hi * 8;             // K-slice offset per ISA layout
    const int rowBase = blockIdx.x * 16;
    const int n0      = blockIdx.y * 64;
    const int mat     = blockIdx.z;

    const unsigned short* W    = Wb + (size_t)mat * 1024u * 1024u;
    const unsigned short* aRow = X + (size_t)(rowBase + r) * 1024u;

    const v8f Z = {0.f, 0.f, 0.f, 0.f, 0.f, 0.f, 0.f, 0.f};
    v8f acc[4] = {Z, Z, Z, Z};

    for (int k0 = 0; k0 < 1024; k0 += 32) {
        Frag a;
        a.u[0] = *(const uint4*)(aRow + k0 + koff);
        a.u[1] = *(const uint4*)(aRow + k0 + 16 + koff);
#pragma unroll
        for (int nt = 0; nt < 4; ++nt) {
            const unsigned short* wRow = W + (size_t)(n0 + nt * 16 + r) * 1024u + k0;
            Frag bfr;
            bfr.u[0] = *(const uint4*)(wRow + koff);
            bfr.u[1] = *(const uint4*)(wRow + 16 + koff);
            acc[nt] = __builtin_amdgcn_wmma_f32_16x16x32_bf16(
                false, a.v, false, bfr.v, (short)0, acc[nt], false, false);
        }
    }

    const float* bias = (mat == 0) ? bq : (mat == 1) ? bk : bv;

    if (mat < 2) {
        unsigned short* O = (mat == 0) ? Qb : Kb;
#pragma unroll
        for (int nt = 0; nt < 4; ++nt) {
            const int c = n0 + nt * 16 + r;
            const float bb = bias[c];
#pragma unroll
            for (int v = 0; v < 8; ++v) {
                const int row = rowBase + hi * 8 + v;
                O[(size_t)row * 1024u + c] = f2bf(acc[nt][v] + bb);
            }
        }
    } else {
        // V transposed: Vt[((b*16 + h)*64 + dv)*1280 + key], keys contiguous
        const int b       = rowBase / 1280;
        const int keyBase = rowBase - b * 1280 + hi * 8;
#pragma unroll
        for (int nt = 0; nt < 4; ++nt) {
            const int c  = n0 + nt * 16 + r;
            const int h  = c >> 6;
            const int dv = c & 63;
            const float bb = bias[c];
            union { unsigned short s[8]; uint4 u; } pk;
#pragma unroll
            for (int v = 0; v < 8; ++v) pk.s[v] = f2bf(acc[nt][v] + bb);
            size_t off = ((size_t)(b * 16 + h) * 64u + dv) * 1280u + keyBase;
            *(uint4*)(Vt + off) = pk.u;   // 8 consecutive keys -> one b128 store
        }
    }
}

// ---------------------------------------------------------------------------
// Flash attention: one wave per (b, h, 16-query tile); nk=1280 in 32-key
// chunks. Scores via 2x wmma(Q,K), online softmax with shfl_xor row
// reductions, P reshaped D-layout -> A-layout through a tiny LDS tile,
// output via 4x wmma(P, Vt).
//   grid = (64 q-tiles, 16 heads, 8 batch), block = 32
// ---------------------------------------------------------------------------
__global__ __launch_bounds__(32) void attn_kernel(
    const unsigned short* __restrict__ Qb,
    const unsigned short* __restrict__ Kb,
    const unsigned short* __restrict__ Vt,
    float* __restrict__ out)
{
    __shared__ unsigned short lp[16 * 40];   // 16 rows x 32 P vals, pad to 40

    const int lane = threadIdx.x;
    const int r    = lane & 15;
    const int hi   = lane >> 4;
    const int koff = hi * 8;
    const int q0 = blockIdx.x * 16;
    const int h  = blockIdx.y;
    const int b  = blockIdx.z;

    // Q A-fragments (row q0+r, d = 0..63) — held for the whole kv loop
    const unsigned short* qrow = Qb + ((size_t)(b * 1280 + q0 + r) * 1024u + h * 64);
    Frag qa0, qa1;
    qa0.u[0] = *(const uint4*)(qrow + koff);
    qa0.u[1] = *(const uint4*)(qrow + 16 + koff);
    qa1.u[0] = *(const uint4*)(qrow + 32 + koff);
    qa1.u[1] = *(const uint4*)(qrow + 48 + koff);

    const v8f Z = {0.f, 0.f, 0.f, 0.f, 0.f, 0.f, 0.f, 0.f};
    v8f o0 = Z, o1 = Z, o2 = Z, o3 = Z;
    float m_i[8], l_i[8];
#pragma unroll
    for (int v = 0; v < 8; ++v) { m_i[v] = -__builtin_inff(); l_i[v] = 0.f; }

    const unsigned short* vbase = Vt + ((size_t)(b * 16 + h) * 64u) * 1280u;

    for (int kc = 0; kc < 1280; kc += 32) {
        // ---- scores: two 16x16 tiles over d=64 (2 wmma each) ----
        v8f s0 = Z, s1 = Z;
        {
            const unsigned short* krow = Kb + ((size_t)(b * 1280 + kc + r) * 1024u + h * 64);
            Frag kb0, kb1;
            kb0.u[0] = *(const uint4*)(krow + koff);
            kb0.u[1] = *(const uint4*)(krow + 16 + koff);
            kb1.u[0] = *(const uint4*)(krow + 32 + koff);
            kb1.u[1] = *(const uint4*)(krow + 48 + koff);
            s0 = __builtin_amdgcn_wmma_f32_16x16x32_bf16(false, qa0.v, false, kb0.v, (short)0, s0, false, false);
            s0 = __builtin_amdgcn_wmma_f32_16x16x32_bf16(false, qa1.v, false, kb1.v, (short)0, s0, false, false);
            const unsigned short* krow2 = krow + (size_t)16 * 1024u;
            kb0.u[0] = *(const uint4*)(krow2 + koff);
            kb0.u[1] = *(const uint4*)(krow2 + 16 + koff);
            kb1.u[0] = *(const uint4*)(krow2 + 32 + koff);
            kb1.u[1] = *(const uint4*)(krow2 + 48 + koff);
            s1 = __builtin_amdgcn_wmma_f32_16x16x32_bf16(false, qa0.v, false, kb0.v, (short)0, s1, false, false);
            s1 = __builtin_amdgcn_wmma_f32_16x16x32_bf16(false, qa1.v, false, kb1.v, (short)0, s1, false, false);
        }

        // ---- online softmax (rows live across the 16-lane half-groups) ----
#pragma unroll
        for (int v = 0; v < 8; ++v) {
            float a = s0[v] * 0.125f;          // 1/sqrt(64)
            float c = s1[v] * 0.125f;
            float t = fmaxf(a, c);
            t = fmaxf(t, __shfl_xor(t, 1, 32));
            t = fmaxf(t, __shfl_xor(t, 2, 32));
            t = fmaxf(t, __shfl_xor(t, 4, 32));
            t = fmaxf(t, __shfl_xor(t, 8, 32));
            const float mnew = fmaxf(m_i[v], t);
            const float sc   = __expf(m_i[v] - mnew);
            const float p0 = __expf(a - mnew);
            const float p1 = __expf(c - mnew);
            float rs = p0 + p1;
            rs += __shfl_xor(rs, 1, 32);
            rs += __shfl_xor(rs, 2, 32);
            rs += __shfl_xor(rs, 4, 32);
            rs += __shfl_xor(rs, 8, 32);
            l_i[v] = l_i[v] * sc + rs;
            m_i[v] = mnew;
            o0[v] *= sc; o1[v] *= sc; o2[v] *= sc; o3[v] *= sc;
            lp[(hi * 8 + v) * 40 + r]      = f2bf(p0);
            lp[(hi * 8 + v) * 40 + 16 + r] = f2bf(p1);
        }
        asm volatile("s_wait_dscnt 0" ::: "memory");

        // ---- P: D-layout -> A-layout via LDS ----
        Frag pa;
        pa.u[0] = *(const uint4*)(lp + r * 40 + koff);
        pa.u[1] = *(const uint4*)(lp + r * 40 + 16 + koff);

        // ---- out += P @ V (Vt keys contiguous -> b128 loads) ----
        {
            Frag vb;
            const unsigned short* vr0 = vbase + (size_t)(0 * 16 + r) * 1280u + kc;
            vb.u[0] = *(const uint4*)(vr0 + koff);
            vb.u[1] = *(const uint4*)(vr0 + 16 + koff);
            o0 = __builtin_amdgcn_wmma_f32_16x16x32_bf16(false, pa.v, false, vb.v, (short)0, o0, false, false);
            const unsigned short* vr1 = vbase + (size_t)(1 * 16 + r) * 1280u + kc;
            vb.u[0] = *(const uint4*)(vr1 + koff);
            vb.u[1] = *(const uint4*)(vr1 + 16 + koff);
            o1 = __builtin_amdgcn_wmma_f32_16x16x32_bf16(false, pa.v, false, vb.v, (short)0, o1, false, false);
            const unsigned short* vr2 = vbase + (size_t)(2 * 16 + r) * 1280u + kc;
            vb.u[0] = *(const uint4*)(vr2 + koff);
            vb.u[1] = *(const uint4*)(vr2 + 16 + koff);
            o2 = __builtin_amdgcn_wmma_f32_16x16x32_bf16(false, pa.v, false, vb.v, (short)0, o2, false, false);
            const unsigned short* vr3 = vbase + (size_t)(3 * 16 + r) * 1280u + kc;
            vb.u[0] = *(const uint4*)(vr3 + koff);
            vb.u[1] = *(const uint4*)(vr3 + 16 + koff);
            o3 = __builtin_amdgcn_wmma_f32_16x16x32_bf16(false, pa.v, false, vb.v, (short)0, o3, false, false);
        }
    }

    // ---- epilogue: normalize and write f32 out[b, q, h*64 + dv] ----
#pragma unroll
    for (int v = 0; v < 8; ++v) {
        const float inv = 1.0f / l_i[v];
        const size_t rowOff = ((size_t)(b * 1024 + q0 + hi * 8 + v)) * 1024u + h * 64;
        out[rowOff + 0 * 16 + r] = o0[v] * inv;
        out[rowOff + 1 * 16 + r] = o1[v] * inv;
        out[rowOff + 2 * 16 + r] = o2[v] * inv;
        out[rowOff + 3 * 16 + r] = o3[v] * inv;
    }
}

// ---------------------------------------------------------------------------
// Launch
// ---------------------------------------------------------------------------
extern "C" void kernel_launch(void* const* d_in, const int* in_sizes, int n_in,
                              void* d_out, int out_size, void* d_ws, size_t ws_size,
                              hipStream_t stream) {
    (void)in_sizes; (void)n_in; (void)out_size; (void)ws_size;
    const float* hid = (const float*)d_in[0];
    const float* ctx = (const float*)d_in[1];
    const float* Wq  = (const float*)d_in[2];
    const float* bq  = (const float*)d_in[3];
    const float* Wk  = (const float*)d_in[4];
    const float* bk  = (const float*)d_in[5];
    const float* Wv  = (const float*)d_in[6];
    const float* bv  = (const float*)d_in[7];

    unsigned short* ws = (unsigned short*)d_ws;
    const size_t NX = (size_t)8 * 1280 * 1024;      // 10,485,760
    const size_t NW = (size_t)3 * 1024 * 1024;      //  3,145,728
    unsigned short* X  = ws;
    unsigned short* Wb = X + NX;
    unsigned short* Qb = Wb + NW;
    unsigned short* Kb = Qb + NX;
    unsigned short* Vt = Kb + NX;                   // total ~90 MB bf16 scratch

    pack_x_kernel<<<(unsigned)(NX / 256), 256, 0, stream>>>(hid, ctx, X);
    pack_w_kernel<<<(unsigned)(NW / 256), 256, 0, stream>>>(Wq, Wk, Wv, Wb);
    proj_kernel<<<dim3(640, 16, 3), 32, 0, stream>>>(X, Wb, bq, bk, bv, Qb, Kb, Vt);
    attn_kernel<<<dim3(64, 16, 8), 32, 0, stream>>>(Qb, Kb, Vt, (float*)d_out);
}
